// PreNormOrderedReductionSA_85341000171865
// MI455X (gfx1250) — matrix operand order repaired and checked by reading, not compiled, not run on hardware
//
#include <hip/hip_runtime.h>
#include <hip/hip_bf16.h>

// ---------------------------------------------------------------------------
// PreNorm reduction self-attention for MI455X (gfx1250), wave32 + WMMA bf16.
// B=4, H=W=64, D=256, NH=8, HD=32, R=2.  N_q=4096/batch, N_k=1024/batch.
// Attention map output (512 MB) is streamed with non-temporal stores so it
// does not evict the K/V/activation working set from the 192 MB L2.
// ---------------------------------------------------------------------------

typedef __attribute__((ext_vector_type(16))) __bf16 v16bf;
typedef __attribute__((ext_vector_type(8)))  __bf16 v8bf;
typedef __attribute__((ext_vector_type(8)))  float  v8f;

#define NB   4
#define NHW  4096          // H*W
#define NRED 1024          // (H/2)*(W/2)
#define DD   256
#define NHEADS 8
#define SCALE 0.17677669529663687f   // 1/sqrt(32)
#define LNEPS 1e-5f

__device__ __forceinline__ v16bf cat8(v8bf a, v8bf b) {
    v16bf r;
#pragma unroll
    for (int i = 0; i < 8; ++i) { r[i] = a[i]; r[i + 8] = b[i]; }
    return r;
}

__device__ __forceinline__ v8f wmma_bf16(v16bf a, v16bf b, v8f c) {
    return __builtin_amdgcn_wmma_f32_16x16x32_bf16(false, a, false, b,
                                                   (short)0, c, false, false);
}

// ---------------------------------------------------------------------------
// LayerNorm over last dim (256), one wave per row, 8 elems/lane.
// ---------------------------------------------------------------------------
__global__ __launch_bounds__(32)
void ln_kernel(const float* __restrict__ in, const float* __restrict__ g,
               const float* __restrict__ bta, __bf16* __restrict__ out) {
    const int row  = blockIdx.x;
    const int lane = threadIdx.x;
    const float* p = in + (size_t)row * DD;
    float x[8]; float s = 0.f;
#pragma unroll
    for (int i = 0; i < 8; ++i) { x[i] = p[i * 32 + lane]; s += x[i]; }
#pragma unroll
    for (int m = 16; m >= 1; m >>= 1) s += __shfl_xor(s, m, 32);
    const float mean = s * (1.f / 256.f);
    float vs = 0.f;
#pragma unroll
    for (int i = 0; i < 8; ++i) { float d = x[i] - mean; vs += d * d; }
#pragma unroll
    for (int m = 16; m >= 1; m >>= 1) vs += __shfl_xor(vs, m, 32);
    const float inv = rsqrtf(vs * (1.f / 256.f) + LNEPS);
    __bf16* o = out + (size_t)row * DD;
#pragma unroll
    for (int i = 0; i < 8; ++i) {
        int c = i * 32 + lane;
        o[c] = (__bf16)((x[i] - mean) * inv * g[c] + bta[c]);
    }
}

// ---------------------------------------------------------------------------
// 2x2 mean pool of raw x -> bf16 x_red  [4096 rows x 256]
// ---------------------------------------------------------------------------
__global__ __launch_bounds__(256)
void pool_kernel(const float* __restrict__ x, __bf16* __restrict__ xr) {
    const int idx = blockIdx.x * 256 + threadIdx.x;   // 4096*256 total
    const int d = idx & 255;
    const int r = idx >> 8;                            // b*1024 + i*32 + j
    const int j = r & 31, i = (r >> 5) & 31, b = r >> 10;
    const size_t base = ((size_t)b * 64 * 64 + (size_t)(2 * i) * 64 + 2 * j) * DD + d;
    float s = x[base] + x[base + DD] + x[base + 64 * DD] + x[base + 64 * DD + DD];
    xr[idx] = (__bf16)(s * 0.25f);
}

// ---------------------------------------------------------------------------
// Pack a 256x256 f32 weight (K x N) into WMMA B-fragment order (bf16).
// Layout: frag (nt*8+ks), lane, 16 elems.  grid=128, block=32.
// B frag: lane -> n = nt*16+(lane&15);  k = ks*32 + (lane>=16?16:0) + e.
// ---------------------------------------------------------------------------
__global__ __launch_bounds__(32)
void packw_kernel(const float* __restrict__ W, __bf16* __restrict__ Wp) {
    const int lane = threadIdx.x;
    const int nt = blockIdx.x >> 3, ks = blockIdx.x & 7;
    const int n  = nt * 16 + (lane & 15);
    const int k0 = ks * 32 + ((lane >> 4) << 4);
    __bf16* o = Wp + ((size_t)blockIdx.x * 32 + lane) * 16;
#pragma unroll
    for (int e = 0; e < 16; ++e) o[e] = (__bf16)W[(size_t)(k0 + e) * DD + n];
}

// ---------------------------------------------------------------------------
// Pack V into per-(b,h) B-fragment order for the P@V WMMA.
// grid = B*NH*32*2, block = 32.  blockIdx: [bh(5b)][kstep(5b)][nt(1b)]
// ---------------------------------------------------------------------------
__global__ __launch_bounds__(32)
void packv_kernel(const __bf16* __restrict__ v_bf, __bf16* __restrict__ Vp) {
    const int lane  = threadIdx.x;
    const int nt    = blockIdx.x & 1;
    const int kstep = (blockIdx.x >> 1) & 31;
    const int bh    = blockIdx.x >> 6;
    const int b = bh >> 3, h = bh & 7;
    const int d   = nt * 16 + (lane & 15);
    const int kk0 = kstep * 32 + ((lane >> 4) << 4);
    __bf16* o = Vp + ((size_t)blockIdx.x * 32 + lane) * 16;
#pragma unroll
    for (int e = 0; e < 16; ++e) {
        int key = kk0 + e;
        o[e] = v_bf[((size_t)(b * NRED + key)) * DD + h * 32 + d];
    }
}

// ---------------------------------------------------------------------------
// bf16 WMMA GEMM: C[M,256] = A_bf[M,256] @ Wp(+bias)(+residual).
// 16 waves / block, wave w owns output column-tile w; blockIdx.x = m-tile.
// ---------------------------------------------------------------------------
__global__ __launch_bounds__(512)
void gemm_bf16_kernel(const __bf16* __restrict__ A, const __bf16* __restrict__ Wp,
                      const float* __restrict__ bias, const float* __restrict__ res,
                      float* __restrict__ outF, __bf16* __restrict__ outB) {
    const int mt   = blockIdx.x;
    const int wv   = threadIdx.x >> 5;     // n-tile 0..15
    const int lane = threadIdx.x & 31;
    const int hi   = lane >> 4;
    const int l15  = lane & 15;
    const int rowA = mt * 16 + l15;

    v8f acc = {};
#pragma unroll
    for (int ks = 0; ks < 8; ++ks) {
        const __bf16* ap = A + (size_t)rowA * DD + ks * 32 + hi * 16;
        v16bf a = cat8(*(const v8bf*)ap, *(const v8bf*)(ap + 8));
        v16bf bf = *(const v16bf*)(Wp + ((size_t)(wv * 8 + ks) * 32 + lane) * 16);
        acc = wmma_bf16(a, bf, acc);
    }
    const int col = wv * 16 + l15;
    const float bb = bias ? bias[col] : 0.f;
#pragma unroll
    for (int r = 0; r < 8; ++r) {
        const int rowO = mt * 16 + r + hi * 8;
        const size_t idx = (size_t)rowO * DD + col;
        float vo = acc[r] + bb;
        if (res)  vo += res[idx];
        if (outF) outF[idx] = vo;
        if (outB) outB[idx] = (__bf16)vo;
    }
}

// ---------------------------------------------------------------------------
// Attention: one wave per (b, h, 16-query tile).  Three WMMA sweeps over the
// 64 key tiles (row-max; exp-sum + P@V via LDS transpose; normalized store).
// Only ONE pass writes the 512MB attn output, with non-temporal stores.
// ---------------------------------------------------------------------------
__global__ __launch_bounds__(32)
void attn_kernel(const __bf16* __restrict__ q_bf, const __bf16* __restrict__ k_bf,
                 const __bf16* __restrict__ Vp, __bf16* __restrict__ o_bf,
                 float* __restrict__ attn) {
    __shared__ __align__(16) __bf16 Pbuf[16 * 32];   // 1 KB / wave

    const int lane = threadIdx.x;
    const int hi   = lane >> 4;
    const int l15  = lane & 15;
    const int qt = blockIdx.x & 255;
    const int h  = (blockIdx.x >> 8) & 7;
    const int b  = blockIdx.x >> 11;
    const int qbase = qt * 16;

    // Q fragment (A layout), resident for all sweeps
    const __bf16* qp = q_bf + ((size_t)(b * NHW + qbase + l15)) * DD + h * 32 + hi * 16;
    const v16bf qa = cat8(*(const v8bf*)qp, *(const v8bf*)(qp + 8));

    const __bf16* kb_base = k_bf + ((size_t)b * NRED) * DD + h * 32 + hi * 16;

    // ---- sweep 1: row max -------------------------------------------------
    float m[8];
#pragma unroll
    for (int r = 0; r < 8; ++r) m[r] = -3.0e38f;
    for (int kt = 0; kt < 64; ++kt) {
        v16bf kb = *(const v16bf*)(kb_base + (size_t)(kt * 16 + l15) * DD);
        v8f s = {};
        s = wmma_bf16(qa, kb, s);
#pragma unroll
        for (int r = 0; r < 8; ++r) m[r] = fmaxf(m[r], s[r] * SCALE);
    }
#pragma unroll
    for (int r = 0; r < 8; ++r)
#pragma unroll
        for (int msk = 1; msk < 16; msk <<= 1)
            m[r] = fmaxf(m[r], __shfl_xor(m[r], msk, 32));

    // ---- sweep 2: exp-sums + O = P@V (unnormalized, scaled at end) --------
    float l[8];
#pragma unroll
    for (int r = 0; r < 8; ++r) l[r] = 0.f;
    v8f acc0 = {}, acc1 = {};
    const __bf16* vfrag = Vp + ((size_t)(b * NHEADS + h) * 32) * 2 * 512 + (size_t)lane * 16;

    for (int kt2 = 0; kt2 < 32; ++kt2) {
#pragma unroll
        for (int sub = 0; sub < 2; ++sub) {
            const int kt = kt2 * 2 + sub;
            v16bf kb = *(const v16bf*)(kb_base + (size_t)(kt * 16 + l15) * DD);
            v8f s = {};
            s = wmma_bf16(qa, kb, s);
#pragma unroll
            for (int r = 0; r < 8; ++r) {
                float p = __expf(s[r] * SCALE - m[r]);
                l[r] += p;
                Pbuf[(r + hi * 8) * 32 + sub * 16 + l15] = (__bf16)p;  // D -> row-major
            }
        }
        __syncthreads();   // single-wave WG: lowers to S_NOP + DS waits
        // reload as A fragment
        const __bf16* pp = Pbuf + (size_t)l15 * 32 + hi * 16;
        v16bf pa = cat8(*(const v8bf*)pp, *(const v8bf*)(pp + 8));
        v16bf vb0 = *(const v16bf*)(vfrag + (size_t)kt2 * 2 * 512);
        v16bf vb1 = *(const v16bf*)(vfrag + ((size_t)kt2 * 2 + 1) * 512);
        acc0 = wmma_bf16(pa, vb0, acc0);
        acc1 = wmma_bf16(pa, vb1, acc1);
        __syncthreads();
    }
#pragma unroll
    for (int r = 0; r < 8; ++r)
#pragma unroll
        for (int msk = 1; msk < 16; msk <<= 1)
            l[r] += __shfl_xor(l[r], msk, 32);
    float rl[8];
#pragma unroll
    for (int r = 0; r < 8; ++r) rl[r] = 1.f / l[r];

    // store per-head output (bf16, feeds final projection GEMM)
#pragma unroll
    for (int r = 0; r < 8; ++r) {
        const size_t orow = (size_t)(b * NHW + qbase + r + hi * 8) * DD + h * 32;
        o_bf[orow + l15]      = (__bf16)(acc0[r] * rl[r]);
        o_bf[orow + 16 + l15] = (__bf16)(acc1[r] * rl[r]);
    }

    // ---- sweep 3: single non-temporal write of normalized attention -------
    float* ap = attn + ((size_t)(b * NHEADS + h) * NHW + qbase) * NRED;
    for (int kt = 0; kt < 64; ++kt) {
        v16bf kb = *(const v16bf*)(kb_base + (size_t)(kt * 16 + l15) * DD);
        v8f s = {};
        s = wmma_bf16(qa, kb, s);
#pragma unroll
        for (int r = 0; r < 8; ++r) {
            float p = __expf(s[r] * SCALE - m[r]) * rl[r];
            __builtin_nontemporal_store(
                p, ap + (size_t)(r + hi * 8) * NRED + kt * 16 + l15);
        }
    }
}

// ---------------------------------------------------------------------------
extern "C" void kernel_launch(void* const* d_in, const int* in_sizes, int n_in,
                              void* d_out, int out_size, void* d_ws, size_t ws_size,
                              hipStream_t stream) {
    const float* x     = (const float*)d_in[0];
    const float* ln1_g = (const float*)d_in[2];
    const float* ln1_b = (const float*)d_in[3];
    const float* Wq    = (const float*)d_in[4];
    const float* bq    = (const float*)d_in[5];
    const float* Wk    = (const float*)d_in[6];
    const float* bk    = (const float*)d_in[7];
    const float* Wv    = (const float*)d_in[8];
    const float* bv    = (const float*)d_in[9];
    const float* Wmean = (const float*)d_in[10];
    const float* ln2_g = (const float*)d_in[11];
    const float* ln2_b = (const float*)d_in[12];
    const float* Wo    = (const float*)d_in[13];
    const float* bo    = (const float*)d_in[14];

    float* outF = (float*)d_out;                          // [4,64,64,256]
    float* attn = outF + (size_t)NB * NHW * DD;           // [4,8,4096,1024]

    // workspace carve (~41 MB)
    size_t off = 0;
    auto carve = [&](size_t bytes) {
        char* p = (char*)d_ws + off;
        off += (bytes + 255) & ~(size_t)255;
        return p;
    };
    __bf16* xn_bf  = (__bf16*)carve((size_t)NB * NHW * DD * 2);   // LN1(x)
    __bf16* q_bf   = (__bf16*)carve((size_t)NB * NHW * DD * 2);
    __bf16* xr_bf  = (__bf16*)carve((size_t)NB * NRED * DD * 2);  // pooled x
    float*  xrm_f  = (float*) carve((size_t)NB * NRED * DD * 4);  // x_red@Wmean
    __bf16* xrn_bf = (__bf16*)carve((size_t)NB * NRED * DD * 2);  // LN2
    __bf16* k_bf   = (__bf16*)carve((size_t)NB * NRED * DD * 2);
    __bf16* v_bf   = (__bf16*)carve((size_t)NB * NRED * DD * 2);
    __bf16* o_bf   = (__bf16*)carve((size_t)NB * NHW * DD * 2);   // attn out heads
    __bf16* Vp     = (__bf16*)carve((size_t)NB * NHEADS * NRED * 32 * 2);
    __bf16* Wq_p   = (__bf16*)carve((size_t)DD * DD * 2);
    __bf16* Wk_p   = (__bf16*)carve((size_t)DD * DD * 2);
    __bf16* Wv_p   = (__bf16*)carve((size_t)DD * DD * 2);
    __bf16* Wm_p   = (__bf16*)carve((size_t)DD * DD * 2);
    __bf16* Wo_p   = (__bf16*)carve((size_t)DD * DD * 2);

    // 1) pack weights into WMMA B-fragment layout
    packw_kernel<<<128, 32, 0, stream>>>(Wq, Wq_p);
    packw_kernel<<<128, 32, 0, stream>>>(Wk, Wk_p);
    packw_kernel<<<128, 32, 0, stream>>>(Wv, Wv_p);
    packw_kernel<<<128, 32, 0, stream>>>(Wmean, Wm_p);
    packw_kernel<<<128, 32, 0, stream>>>(Wo, Wo_p);

    // 2) LN1(x) -> bf16
    ln_kernel<<<NB * NHW, 32, 0, stream>>>(x, ln1_g, ln1_b, xn_bf);
    // 3) 2x2 mean pool of raw x
    pool_kernel<<<(NB * NRED * DD) / 256, 256, 0, stream>>>(x, xr_bf);
    // 4) q = LN1(x) @ Wq + bq    (bf16 out)
    gemm_bf16_kernel<<<(NB * NHW) / 16, 512, 0, stream>>>(xn_bf, Wq_p, bq, nullptr,
                                                          nullptr, q_bf);
    // 5) x_red @ Wmean  (f32 out for LN2)
    gemm_bf16_kernel<<<(NB * NRED) / 16, 512, 0, stream>>>(xr_bf, Wm_p, nullptr, nullptr,
                                                           xrm_f, nullptr);
    // 6) LN2
    ln_kernel<<<NB * NRED, 32, 0, stream>>>(xrm_f, ln2_g, ln2_b, xrn_bf);
    // 7) k, v projections (bf16 out)
    gemm_bf16_kernel<<<(NB * NRED) / 16, 512, 0, stream>>>(xrn_bf, Wk_p, bk, nullptr,
                                                           nullptr, k_bf);
    gemm_bf16_kernel<<<(NB * NRED) / 16, 512, 0, stream>>>(xrn_bf, Wv_p, bv, nullptr,
                                                           nullptr, v_bf);
    // 8) pack V per (b,h) into B-fragment layout
    packv_kernel<<<NB * NHEADS * 32 * 2, 32, 0, stream>>>(v_bf, Vp);
    // 9) attention (writes attn output + per-head O)
    attn_kernel<<<NB * NHEADS * (NHW / 16), 32, 0, stream>>>(q_bf, k_bf, Vp, o_bf, attn);
    // 10) out = O @ Wo + bo + x  (f32 -> d_out)
    gemm_bf16_kernel<<<(NB * NHW) / 16, 512, 0, stream>>>(o_bf, Wo_p, bo, x,
                                                          outF, nullptr);
}
